// NonLocalBlock_16363825398150
// MI455X (gfx1250) — matrix operand order repaired and checked by reading, not compiled, not run on hardware
//
#include <hip/hip_runtime.h>
#include <hip/hip_bf16.h>

typedef __bf16 bf16;
typedef __attribute__((ext_vector_type(16))) __bf16 v16bf;
typedef __attribute__((ext_vector_type(8)))  float   v8f;
typedef __attribute__((ext_vector_type(4)))  int     v4i;

#define BN 8
#define HH 56
#define WW 56
#define CC 256
#define CI 128
#define NN (HH * WW)   // 3136
#define NR (BN * NN)   // 25088 total rows
#define MM (NN / 2)    // 1568 pooled rows per batch
#define MR (BN * MM)   // 12544 total pooled rows

#if defined(__has_builtin)
#if __has_builtin(__builtin_amdgcn_global_load_async_to_lds_b128)
#define HAVE_ASYNC_COPY 1
#endif
#if __has_builtin(__builtin_amdgcn_s_wait_asynccnt)
#define HAVE_ASYNC_WAIT 1
#endif
#endif

typedef __attribute__((address_space(1))) v4i gv4i;  // global 16B chunk
typedef __attribute__((address_space(3))) v4i lv4i;  // LDS 16B chunk

// 16-byte global -> LDS copy. Async (ASYNCcnt-tracked, no VGPR round trip) when
// the toolchain exposes the gfx1250 builtin; plain b128 load+ds_store otherwise.
static __device__ __forceinline__ void copy16_g2l(const bf16* gsrc, bf16* ldst) {
#if defined(HAVE_ASYNC_COPY)
  __builtin_amdgcn_global_load_async_to_lds_b128((gv4i*)gsrc, (lv4i*)ldst, 0, 0);
#else
  *reinterpret_cast<uint4*>(ldst) = *reinterpret_cast<const uint4*>(gsrc);
#endif
}

static __device__ __forceinline__ void staging_wait() {
#if defined(HAVE_ASYNC_COPY)
#if defined(HAVE_ASYNC_WAIT)
  __builtin_amdgcn_s_wait_asynccnt(0);
#else
  asm volatile("s_wait_asynccnt 0x0" ::: "memory");
#endif
#endif
  asm volatile("s_wait_dscnt 0x0" ::: "memory");
}

static __device__ __forceinline__ v8f wmma_bf16(v16bf a, v16bf b, v8f c) {
  return __builtin_amdgcn_wmma_f32_16x16x32_bf16(false, a, false, b, (short)0, c,
                                                 false, false);
}

// A-matrix fragment (16x32 bf16), row-major tile with leading dim lda.
// Layout: M = lane&15; VGPR v holds K = 2v + 16h (+1), h = lane>>4 (matches B).
// (32 contiguous bytes per lane.)
static __device__ __forceinline__ v16bf load_a_frag(const bf16* At, int lda,
                                                    int kbase, int lane) {
  const int m = lane & 15, h = lane >> 4;
  v16bf a;
#pragma unroll
  for (int v = 0; v < 8; ++v) {
    int kp = kbase + 2 * v + 16 * h;
    a[2 * v]     = At[m * lda + kp];
    a[2 * v + 1] = At[m * lda + kp + 1];
  }
  return a;
}

// B-matrix fragment (32x16) from a K-major [k][n] tile, leading dim ldb.
// Layout used here (contiguous k, same as A): lanes 0-15 hold K=0-15 (VGPR v -> K=2v,2v+1), lanes 16-31 K=16-31.
static __device__ __forceinline__ v16bf load_b_frag(const bf16* Bt, int ldb,
                                                    int col0, int kbase, int lane) {
  const int n = lane & 15, h = lane >> 4;
  v16bf b;
#pragma unroll
  for (int v = 0; v < 8; ++v) {
    int k = kbase + 2 * v + 16 * h;
    b[2 * v]     = Bt[k * ldb + col0 + n];
    b[2 * v + 1] = Bt[(k + 1) * ldb + col0 + n];
  }
  return b;
}

// B fragment where logical B[k][j] = T[j][k] (N-major / transposed source),
// T row-major with leading dim ldt. 32 contiguous bytes per lane.
static __device__ __forceinline__ v16bf load_bT_frag(const bf16* Tt, int ldt,
                                                     int row0, int kbase, int lane) {
  const int n = lane & 15, h = lane >> 4;
  v16bf b;
#pragma unroll
  for (int v = 0; v < 8; ++v) {
    int k = kbase + 2 * v + 16 * h;
    b[2 * v]     = Tt[(row0 + n) * ldt + k];
    b[2 * v + 1] = Tt[(row0 + n) * ldt + k + 1];
  }
  return b;
}

// ------------- Kernel 0: one-time weight convert + transpose to bf16 -------------
// dst[n*K + k] = (bf16) src[k*Nc + n]   (dst is N-major so K-pairs are contiguous)
__global__ __launch_bounds__(256) void wtrans_kernel(const float* __restrict__ src,
                                                     bf16* __restrict__ dst, int K,
                                                     int Nc) {
  int i = blockIdx.x * blockDim.x + threadIdx.x;
  if (i < K * Nc) {
    int n = i / K, k = i - n * K;
    dst[i] = (bf16)src[k * Nc + n];
  }
}

// ---------------- Kernel 1: projections x @ {w_theta,w_phi,w_g} ----------------
// One wave per (16-row tile, weight); A-fragments built once, reused for all 8
// Ci column tiles (64 WMMAs per block).
__global__ __launch_bounds__(32) void proj_kernel(
    const float* __restrict__ x, const bf16* __restrict__ wtT,
    const bf16* __restrict__ wpT, const bf16* __restrict__ wgT,
    bf16* __restrict__ theta, bf16* __restrict__ phif, bf16* __restrict__ gfull) {
  __shared__ bf16 Xs[16 * CC];  // 8 KB
  const int lane = threadIdx.x;
  const int rt = blockIdx.x;    // 0..NR/16-1
  const int wsel = blockIdx.y;  // 0..2
  const int r0 = rt * 16;

  const float4* xg = reinterpret_cast<const float4*>(x + (size_t)r0 * CC);
#pragma unroll
  for (int t = 0; t < 32; ++t) {
    int idx = t * 32 + lane;  // 0..1023
    float4 v = xg[idx];
    union { bf16 hx[4]; uint2 u; } pk;
    pk.hx[0] = (bf16)v.x; pk.hx[1] = (bf16)v.y;
    pk.hx[2] = (bf16)v.z; pk.hx[3] = (bf16)v.w;
    reinterpret_cast<uint2*>(Xs)[idx] = pk.u;
  }
  asm volatile("s_wait_dscnt 0x0" ::: "memory");

  const bf16* WT = (wsel == 0) ? wtT : (wsel == 1 ? wpT : wgT);  // [CI][CC]
  bf16* Out = (wsel == 0) ? theta : (wsel == 1 ? phif : gfull);

  v16bf afr[8];
#pragma unroll
  for (int ks = 0; ks < 8; ++ks) afr[ks] = load_a_frag(Xs, CC, ks * 32, lane);

  const int h = lane >> 4, n = lane & 15;
#pragma unroll
  for (int ct = 0; ct < 8; ++ct) {
    v8f acc = {};
#pragma unroll
    for (int ks = 0; ks < 8; ++ks) {
      v16bf b = load_bT_frag(WT, CC, ct * 16, ks * 32, lane);
      acc = wmma_bf16(afr[ks], b, acc);
    }
#pragma unroll
    for (int r = 0; r < 8; ++r) {
      int row = r0 + r + 8 * h;
      Out[(size_t)row * CI + ct * 16 + n] = (bf16)acc[r];
    }
  }
}

// ---------------- Kernel 2: non-overlapping max-pool (pool=2) along N ----------------
__global__ __launch_bounds__(256) void pool_kernel(const bf16* __restrict__ full,
                                                   bf16* __restrict__ pooled,
                                                   int nElems) {
  int i = blockIdx.x * blockDim.x + threadIdx.x;  // over MR*CI
  if (i < nElems) {
    int m = i / CI, c = i - m * CI;
    float a = (float)full[(size_t)(2 * m) * CI + c];
    float b = (float)full[(size_t)(2 * m + 1) * CI + c];
    pooled[i] = (bf16)fmaxf(a, b);
  }
}

// ------------- Kernel 3: flash attention  y = softmax(theta phi^T) g -------------
__global__ __launch_bounds__(32) void attn_kernel(const bf16* __restrict__ theta,
                                                  const bf16* __restrict__ phi,
                                                  const bf16* __restrict__ g,
                                                  bf16* __restrict__ y) {
  __shared__ bf16 Th[16 * CI];   // 4 KB  query tile
  __shared__ bf16 Phs[32 * CI];  // 8 KB  key tile
  __shared__ bf16 Gs[32 * CI];   // 8 KB  value tile
  __shared__ bf16 Ps[16 * 32];   // 1 KB  probabilities (A fragment staging)
  const int lane = threadIdx.x;
  const int batch = blockIdx.y;
  const int nt = blockIdx.x;  // 0..NN/16-1
  const size_t thRow0 = (size_t)batch * NN + nt * 16;

  {  // stage theta tile (256 x 16B chunks, 8 per lane)
    const bf16* src = theta + thRow0 * CI;
#pragma unroll
    for (int t = 0; t < 8; ++t)
      copy16_g2l(src + (t * 32 + lane) * 8, Th + (t * 32 + lane) * 8);
  }
  staging_wait();

  v16bf afr[4];
#pragma unroll
  for (int ks = 0; ks < 4; ++ks) afr[ks] = load_a_frag(Th, CI, ks * 32, lane);

  float mst[8], lst[8];
  v8f yac[8];
#pragma unroll
  for (int r = 0; r < 8; ++r) { mst[r] = -1e30f; lst[r] = 0.0f; }
#pragma unroll
  for (int c8 = 0; c8 < 8; ++c8) { v8f z = {}; yac[c8] = z; }

  const int h = lane >> 4, n = lane & 15;

  for (int jt = 0; jt < MM / 32; ++jt) {  // 49 key/value chunks of 32 rows
    const size_t pRow0 = (size_t)batch * MM + (size_t)jt * 32;
    // Previous iteration's DS fragment reads must drain before LDS is rewritten
    // (async LDS writes are ASYNCcnt-tracked, unordered vs DScnt).
    asm volatile("s_wait_dscnt 0x0" ::: "memory");
    {  // stage phi & g chunks (512 x 16B chunks each, 16 per lane)
      const bf16* sp = phi + pRow0 * CI;
      const bf16* sg = g + pRow0 * CI;
#pragma unroll
      for (int t = 0; t < 16; ++t) {
        copy16_g2l(sp + (t * 32 + lane) * 8, Phs + (t * 32 + lane) * 8);
        copy16_g2l(sg + (t * 32 + lane) * 8, Gs + (t * 32 + lane) * 8);
      }
    }
    // Prefetch next chunk while this one is consumed (global_prefetch_b8).
    if (jt + 1 < MM / 32) {
      const char* np = (const char*)(phi + (pRow0 + 32) * CI);
      const char* ng = (const char*)(g + (pRow0 + 32) * CI);
      __builtin_prefetch(np + lane * 256, 0, 0);
      __builtin_prefetch(np + lane * 256 + 128, 0, 0);
      __builtin_prefetch(ng + lane * 256, 0, 0);
      __builtin_prefetch(ng + lane * 256 + 128, 0, 0);
    }
    staging_wait();

    // S = theta_tile @ phi_chunk^T  -> two 16x16 f32 fragments (cols 0-15, 16-31)
    v8f s0 = {}, s1 = {};
#pragma unroll
    for (int ks = 0; ks < 4; ++ks) {
      v16bf b0 = load_bT_frag(Phs, CI, 0, ks * 32, lane);
      s0 = wmma_bf16(afr[ks], b0, s0);
      v16bf b1 = load_bT_frag(Phs, CI, 16, ks * 32, lane);
      s1 = wmma_bf16(afr[ks], b1, s1);
    }

    // Online softmax per output row (row = r + 8h, cols spread over 16 lanes)
#pragma unroll
    for (int r = 0; r < 8; ++r) {
      float v0 = s0[r], v1 = s1[r];
      float mx = fmaxf(v0, v1);
#pragma unroll
      for (int d = 8; d >= 1; d >>= 1) mx = fmaxf(mx, __shfl_xor(mx, d, 16));
      float mnew = fmaxf(mst[r], mx);
      float p0 = __expf(v0 - mnew);
      float p1 = __expf(v1 - mnew);
      float rs = p0 + p1;
#pragma unroll
      for (int d = 8; d >= 1; d >>= 1) rs += __shfl_xor(rs, d, 16);
      float sc = __expf(mst[r] - mnew);
      lst[r] = lst[r] * sc + rs;
      mst[r] = mnew;
#pragma unroll
      for (int c8 = 0; c8 < 8; ++c8) yac[c8][r] *= sc;
      Ps[(r + 8 * h) * 32 + n] = (bf16)p0;
      Ps[(r + 8 * h) * 32 + n + 16] = (bf16)p1;
    }
    asm volatile("s_wait_dscnt 0x0" ::: "memory");

    // Y += P @ G_chunk   (P is exactly one 16x32 A fragment)
    v16bf pf = load_a_frag(Ps, 32, 0, lane);
#pragma unroll
    for (int c8 = 0; c8 < 8; ++c8) {
      v16bf bg = load_b_frag(Gs, CI, c8 * 16, 0, lane);
      yac[c8] = wmma_bf16(pf, bg, yac[c8]);
    }
    asm volatile("" ::: "memory");
  }

  // Normalize and store y (bf16)
#pragma unroll
  for (int c8 = 0; c8 < 8; ++c8) {
#pragma unroll
    for (int r = 0; r < 8; ++r) {
      float val = yac[c8][r] / lst[r];
      y[(thRow0 + r + 8 * h) * CI + c8 * 16 + n] = (bf16)val;
    }
  }
}

// ---------------- Kernel 4: z = y @ w_final ; out = x + z ----------------
// One wave per 16-row tile; A-fragments reused for all 16 output col tiles.
__global__ __launch_bounds__(32) void final_kernel(const bf16* __restrict__ y,
                                                   const bf16* __restrict__ wfT,
                                                   const float* __restrict__ x,
                                                   float* __restrict__ out) {
  __shared__ bf16 Ys[16 * CI];  // 4 KB
  const int lane = threadIdx.x;
  const int rt = blockIdx.x;  // NR/16
  const int r0 = rt * 16;
  {
    const bf16* src = y + (size_t)r0 * CI;
#pragma unroll
    for (int t = 0; t < 8; ++t)
      copy16_g2l(src + (t * 32 + lane) * 8, Ys + (t * 32 + lane) * 8);
  }
  staging_wait();

  v16bf afr[4];
#pragma unroll
  for (int ks = 0; ks < 4; ++ks) afr[ks] = load_a_frag(Ys, CI, ks * 32, lane);

  const int h = lane >> 4, n = lane & 15;
#pragma unroll
  for (int ct = 0; ct < 16; ++ct) {
    v8f acc = {};
#pragma unroll
    for (int ks = 0; ks < 4; ++ks) {
      v16bf b = load_bT_frag(wfT, CI, ct * 16, ks * 32, lane);
      acc = wmma_bf16(afr[ks], b, acc);
    }
#pragma unroll
    for (int r = 0; r < 8; ++r) {
      size_t idx = (size_t)(r0 + r + 8 * h) * CC + ct * 16 + n;
      out[idx] = x[idx] + acc[r];
    }
  }
}

extern "C" void kernel_launch(void* const* d_in, const int* in_sizes, int n_in,
                              void* d_out, int out_size, void* d_ws, size_t ws_size,
                              hipStream_t stream) {
  const float* x = (const float*)d_in[0];
  const float* w_theta = (const float*)d_in[1];
  const float* w_phi = (const float*)d_in[2];
  const float* w_g = (const float*)d_in[3];
  const float* w_final = (const float*)d_in[4];
  float* out = (float*)d_out;

  char* ws = (char*)d_ws;
  const size_t S1 = (size_t)NR * CI * sizeof(bf16);  // 6.42 MB
  const size_t S2 = (size_t)MR * CI * sizeof(bf16);  // 3.21 MB
  const size_t SW = (size_t)CC * CI * sizeof(bf16);  // 64 KB
  bf16* theta = (bf16*)(ws);
  bf16* phif  = (bf16*)(ws + S1);
  bf16* gfull = (bf16*)(ws + 2 * S1);
  bf16* phi   = (bf16*)(ws + 3 * S1);
  bf16* gp    = (bf16*)(ws + 3 * S1 + S2);
  bf16* y     = (bf16*)(ws + 3 * S1 + 2 * S2);
  char* wbase = ws + 4 * S1 + 2 * S2;
  bf16* wtT = (bf16*)(wbase);           // [CI][CC]
  bf16* wpT = (bf16*)(wbase + SW);      // [CI][CC]
  bf16* wgT = (bf16*)(wbase + 2 * SW);  // [CI][CC]
  bf16* wfT = (bf16*)(wbase + 3 * SW);  // [CC][CI]

  // One-time weight convert+transpose (tiny).
  const int we = CC * CI;
  wtrans_kernel<<<(we + 255) / 256, 256, 0, stream>>>(w_theta, wtT, CC, CI);
  wtrans_kernel<<<(we + 255) / 256, 256, 0, stream>>>(w_phi, wpT, CC, CI);
  wtrans_kernel<<<(we + 255) / 256, 256, 0, stream>>>(w_g, wgT, CC, CI);
  wtrans_kernel<<<(we + 255) / 256, 256, 0, stream>>>(w_final, wfT, CI, CC);

  dim3 g1(NR / 16, 3);
  proj_kernel<<<g1, 32, 0, stream>>>(x, wtT, wpT, wgT, theta, phif, gfull);

  const int pe = MR * CI;
  pool_kernel<<<(pe + 255) / 256, 256, 0, stream>>>(phif, phi, pe);
  pool_kernel<<<(pe + 255) / 256, 256, 0, stream>>>(gfull, gp, pe);

  dim3 g2(NN / 16, BN);
  attn_kernel<<<g2, 32, 0, stream>>>(theta, phi, gp, y);

  dim3 g3(NR / 16);
  final_kernel<<<g3, 32, 0, stream>>>(y, wfT, x, out);
}
